// FastBallQueryPointGrouping_6150393168283
// MI455X (gfx1250) — matrix operand order repaired and checked by reading, not compiled, not run on hardware
//
#include <hip/hip_runtime.h>
#include <hip/hip_bf16.h>

#define Bb 2
#define Nn 16384
#define Mm 4096
#define Kk 32
#define CI 64
#define C0 64
#define C1 128
#define C2 256
#define RADIUS2 0.04f
#define EPSb 1e-5f

typedef __attribute__((ext_vector_type(16))) __bf16 v16bf;
typedef __attribute__((ext_vector_type(8)))  float  v8f;
typedef __attribute__((ext_vector_type(4)))  unsigned int v4u;
typedef __attribute__((ext_vector_type(8)))  int v8i;
typedef __attribute__((ext_vector_type(4)))  int v4i;

#if defined(__has_builtin)
#  if __has_builtin(__builtin_amdgcn_tensor_load_to_lds) && __has_builtin(__builtin_amdgcn_s_wait_tensorcnt)
#    define USE_TDM 1
#  endif
#endif
#ifndef USE_TDM
#  define USE_TDM 0
#endif

// ---------------- workspace layout (float offsets) ----------------
#define WS_Y      0u                      // B*N*C0 = 2097152
#define WS_H1     2097152u                // B*M*C1 = 1048576
#define WS_H2     3145728u                // B*M*C2 = 2097152
#define WS_POOL   5242880u                // B*M*C0 = 524288
#define WS_IDX    5767168u                // B*M*K ints
#define WS_CNT    6029312u                // B*M ints
#define WS_STATS  6037504u                // stats region
// stats sub-offsets (relative to WS_STATS)
#define SF1   0u
#define SF2   64u
#define SMOM  128u
#define S11   160u
#define S12   288u
#define S21   416u
#define S22   672u    // zeroed region = [0, 928)
#define FSC   928u
#define FSH   992u
#define XSC   1056u
#define XSH   1120u
#define SC1   1184u
#define SH1   1312u
#define SC2   1440u
#define SH2   1696u   // end 1952

// =====================================================================
// GEMM (row-major A [R x Kd] @ W [Kd x Nc]) with bf16 WMMA, f32 accum.
// Fully unrolled k-loop (Kd/Nc compile-time). Optional fused
// per-input-channel BN affine + ReLU on A-load. Accumulates per-output-
// channel sum / sum-of-squares for the next BatchNorm.
// One wave computes one 16x16 output tile.
// =====================================================================
template <bool BN_RELU_A, int Kd, int Nc>
__global__ void gemm_bn_stats_kernel(const float* __restrict__ A,
                                     const float* __restrict__ W,
                                     float* __restrict__ Yout,
                                     float* __restrict__ s1,
                                     float* __restrict__ s2,
                                     const float* __restrict__ aScale,
                                     const float* __restrict__ aShift)
{
    __shared__ float lds1[Nc];
    __shared__ float lds2[Nc];
    const int t = threadIdx.x;
    if (t < Nc) { lds1[t] = 0.f; lds2[t] = 0.f; }
    __syncthreads();

    const int lane  = t & 31;
    const int wave  = t >> 5;
    const int gw    = blockIdx.x * 8 + wave;
    const int tileM = gw / (Nc >> 4);
    const int tileN = gw % (Nc >> 4);

    const int half = lane >> 4;         // which 16-lane half
    const int ml   = lane & 15;
    const int row  = tileM * 16 + ml;   // A row for this lane (A-frag)
    const int col  = tileN * 16 + ml;   // output column (B/C/D-frag)

    const float* __restrict__ arow = A + (size_t)row * Kd;
    __builtin_prefetch(arow, 0, 0);

    v8f acc = {};
    #pragma unroll
    for (int kb = 0; kb < Kd; kb += 32) {
        v16bf a, b;
        // A fragment: 16-bit A 16x32 lane layout (ISA 7.12.2)
        #pragma unroll
        for (int v = 0; v < 16; ++v) {
            const int vi = v >> 1, j = v & 1;
            const int kk = kb + ((vi < 4) ? (8 * half + 2 * vi)
                                          : (16 + 8 * half + 2 * (vi - 4))) + j;
            float x = arow[kk];
            if (BN_RELU_A) {
                x = x * aScale[kk] + aShift[kk];
                x = x > 0.f ? x : 0.f;
            }
            a[v] = (__bf16)x;
        }
        // B fragment: 32x16, same K order as A (v<8: 8*half+v, v>=8: 16+8*half+v-8)
        #pragma unroll
        for (int v = 0; v < 16; ++v) {
            const int kk = kb + ((v < 8) ? (8 * half + v) : (16 + 8 * half + (v - 8)));
            b[v] = (__bf16)W[(size_t)kk * Nc + col];
        }
        acc = __builtin_amdgcn_wmma_f32_16x16x32_bf16(false, a, false, b,
                                                      (short)0, acc, false, false);
    }

    // D layout: VGPR v holds row (tileM*16 + 8*half + v), column = col
    float p1 = 0.f, p2 = 0.f;
    #pragma unroll
    for (int v = 0; v < 8; ++v) {
        const int r = tileM * 16 + half * 8 + v;
        const float d = acc[v];
        Yout[(size_t)r * Nc + col] = d;
        p1 += d;
        p2 += d * d;
    }
    atomicAdd(&lds1[col], p1);
    atomicAdd(&lds2[col], p2);
    __syncthreads();
    if (t < Nc) {
        atomicAdd(&s1[t], lds1[t]);
        atomicAdd(&s2[t], lds2[t]);
    }
}

// Fold BN stats into per-channel scale/shift:  y*sc + sh
__global__ void bn_finalize_kernel(const float* __restrict__ s1,
                                   const float* __restrict__ s2,
                                   const float* __restrict__ g,
                                   const float* __restrict__ bta,
                                   float invCount,
                                   float* __restrict__ scale,
                                   float* __restrict__ shift, int C)
{
    const int c = blockIdx.x * blockDim.x + threadIdx.x;
    if (c >= C) return;
    const float mean = s1[c] * invCount;
    const float var  = s2[c] * invCount - mean * mean;
    const float sc   = g[c] * rsqrtf(var + EPSb);
    scale[c] = sc;
    shift[c] = bta[c] - mean * sc;
}

// =====================================================================
// Ball query: one wave per query point, ordered first-K-within-radius
// scan using wave ballots. xyz tiles staged through double-buffered LDS
// via the Tensor Data Mover (async, TENSORcnt-tracked) when available.
// =====================================================================
#define BQ_TILE 256
#define BQ_TBYTES (BQ_TILE * 3 * 4)
#define BQ_TELEMS (BQ_TILE * 3)

#if USE_TDM
__device__ __forceinline__ void tdm_issue_tile(const float* gsrc, float* ldsDst)
{
    const unsigned long long ga = (unsigned long long)(uintptr_t)gsrc;   // byte addr
    const unsigned ldsOff = (unsigned)(uintptr_t)ldsDst;                 // LDS byte offset

    v4u g0;
    g0[0] = 1u;                                        // count=1 (valid), user mode
    g0[1] = ldsOff;                                    // lds_addr
    g0[2] = (unsigned)ga;                              // global_addr[31:0]
    g0[3] = (unsigned)((ga >> 32) & 0x1FFFFFFull) | (2u << 30);  // addr[56:32] | type=2

    v8i g1;
    g1[0] = (int)(2u << 16);                           // wg_mask=0, data_size=4B
    g1[1] = (int)((unsigned)(BQ_TELEMS & 0xFFFF) << 16);   // tensor_dim0[15:0]
    g1[2] = (int)(1u << 16);                           // tensor_dim0 hi=0, tensor_dim1=1
    g1[3] = (int)((unsigned)BQ_TELEMS << 16);          // tile_dim0 = 768
    g1[4] = 1;                                         // tile_dim1 = 1
    g1[5] = BQ_TELEMS;                                 // tensor_dim0_stride
    g1[6] = 0;
    g1[7] = 0;

    const v4i z4 = {0, 0, 0, 0};
    const v8i z8 = {0, 0, 0, 0, 0, 0, 0, 0};
    // 6-arg form (clang-23 / therock-10.0 toolchain)
    __builtin_amdgcn_tensor_load_to_lds(g0, g1, z4, z4, z8, 0);
}
#endif

__global__ void ball_query_kernel(const float* __restrict__ xyz,
                                  const float* __restrict__ new_xyz,
                                  int* __restrict__ idxOut,
                                  int* __restrict__ cntOut)
{
    __shared__ float pts[2][BQ_TELEMS];
    __shared__ int   idxbuf[8][Kk];

    const int t = threadIdx.x, lane = t & 31, wave = t >> 5;
    const int q = blockIdx.x * 8 + wave;          // exact: B*M/8 blocks
    const int b = q / Mm;                         // uniform across block (8 | M)

    const float qx = new_xyz[(size_t)q * 3 + 0];
    const float qy = new_xyz[(size_t)q * 3 + 1];
    const float qz = new_xyz[(size_t)q * 3 + 2];

    idxbuf[wave][lane] = 0;                       // cnt==0 -> idx 0 (matches ref)
    int cnt = 0;                                  // wave-uniform

    const float* __restrict__ base = xyz + (size_t)b * Nn * 3;
    const int NT = Nn / BQ_TILE;

#if USE_TDM
    if (wave == 0) tdm_issue_tile(base, &pts[0][0]);       // prologue: tile 0
#endif
    for (int tile = 0; tile < NT; ++tile) {
#if USE_TDM
        if (wave == 0) {
            if (tile + 1 < NT) {
                tdm_issue_tile(base + (size_t)(tile + 1) * BQ_TELEMS,
                               &pts[(tile + 1) & 1][0]);
                __builtin_amdgcn_s_wait_tensorcnt(1);      // tile's transfer done
            } else {
                __builtin_amdgcn_s_wait_tensorcnt(0);
            }
        }
        __syncthreads();                                   // publish LDS to all waves
        const float* __restrict__ pt = &pts[tile & 1][0];
#else
        __syncthreads();
        {
            const int off = tile * BQ_TELEMS;
            pts[0][t]       = base[off + t];
            pts[0][t + 256] = base[off + t + 256];
            pts[0][t + 512] = base[off + t + 512];
        }
        __syncthreads();
        const float* __restrict__ pt = &pts[0][0];
#endif
        if (cnt < Kk) {
            for (int sub = 0; sub < BQ_TILE / 32; ++sub) {
                if (cnt >= Kk) break;
                const int p  = sub * 32 + lane;
                const float dx = pt[p * 3 + 0] - qx;
                const float dy = pt[p * 3 + 1] - qy;
                const float dz = pt[p * 3 + 2] - qz;
                const bool pass = (dx * dx + dy * dy + dz * dz) < RADIUS2;
#if __has_builtin(__builtin_amdgcn_ballot_w32)
                const unsigned mask = __builtin_amdgcn_ballot_w32(pass);
#else
                const unsigned mask = (unsigned)__ballot(pass);
#endif
                if (pass) {
                    const int pos = cnt + __popc(mask & ((1u << lane) - 1u));
                    if (pos < Kk) idxbuf[wave][pos] = tile * BQ_TILE + p;
                }
                cnt += __popc(mask);
            }
        }
        __syncthreads();                                   // done reading this buffer
    }
    const int cc    = cnt < Kk ? cnt : Kk;
    const int first = idxbuf[wave][0];
    const int myv   = (lane < cc) ? idxbuf[wave][lane] : first;
    idxOut[(size_t)q * Kk + lane] = myv;
    if (lane == 0) cntOut[q] = cc;
}

// =====================================================================
// Second-order moments of grouped_xyz — BN stats of (dxyz @ W_xyz) are a
// quadratic form in these 9 sums, so the (B,M,K,C0) tensor is never
// materialized.
// =====================================================================
__global__ void xyz_moments_kernel(const float* __restrict__ xyz,
                                   const float* __restrict__ new_xyz,
                                   const int* __restrict__ idx,
                                   float* __restrict__ mom)
{
    const int s = blockIdx.x * 256 + threadIdx.x;   // < B*M*K exactly
    const int b = s / (Mm * Kk);
    const int r = s % (Mm * Kk);
    const int m = r / Kk;
    const int i = idx[s];

    const float* qp = new_xyz + ((size_t)b * Mm + m) * 3;
    const float* pp = xyz + ((size_t)b * Nn + i) * 3;
    const float dx = pp[0] - qp[0];
    const float dy = pp[1] - qp[1];
    const float dz = pp[2] - qp[2];

    float v[9] = {dx, dy, dz, dx * dx, dy * dy, dz * dz, dx * dy, dx * dz, dy * dz};
    #pragma unroll
    for (int j = 0; j < 9; ++j) {
        float x = v[j];
        for (int o = 16; o > 0; o >>= 1) x += __shfl_down(x, o, 32);
        v[j] = x;
    }
    __shared__ float lds[9];
    if (threadIdx.x < 9) lds[threadIdx.x] = 0.f;
    __syncthreads();
    if ((threadIdx.x & 31) == 0) {
        #pragma unroll
        for (int j = 0; j < 9; ++j) atomicAdd(&lds[j], v[j]);
    }
    __syncthreads();
    if (threadIdx.x < 9) atomicAdd(&mom[threadIdx.x], lds[threadIdx.x]);
}

__global__ void xyz_bn_finalize_kernel(const float* __restrict__ mom,
                                       const float* __restrict__ Wx,
                                       const float* __restrict__ g,
                                       const float* __restrict__ bta,
                                       float* __restrict__ scale,
                                       float* __restrict__ shift)
{
    const int c = threadIdx.x;   // 64 threads
    const float inv = 1.0f / (float)(Bb * Mm * Kk);
    const float w0 = Wx[c], w1 = Wx[C0 + c], w2 = Wx[2 * C0 + c];
    const float mean = (w0 * mom[0] + w1 * mom[1] + w2 * mom[2]) * inv;
    const float ey2  = (w0 * w0 * mom[3] + w1 * w1 * mom[4] + w2 * w2 * mom[5]
                        + 2.f * (w0 * w1 * mom[6] + w0 * w2 * mom[7] + w1 * w2 * mom[8])) * inv;
    const float var  = ey2 - mean * mean;
    const float sc   = g[c] * rsqrtf(var + EPSb);
    scale[c] = sc;
    shift[c] = bta[c] - mean * sc;
}

// =====================================================================
// Fused gather + xyz-MLP(BN) + add feat(BN) + ReLU + max-pool over K.
// One wave per query; 2 channels per lane; neighbor indices broadcast
// via cross-lane shuffle; feature rows read as coalesced float2.
// =====================================================================
__global__ void group_pool_kernel(const float* __restrict__ xyz,
                                  const float* __restrict__ new_xyz,
                                  const float* __restrict__ Y,
                                  const int* __restrict__ idx,
                                  const float* __restrict__ Wx,
                                  const float* __restrict__ xscale,
                                  const float* __restrict__ xshift,
                                  const float* __restrict__ fscale,
                                  const float* __restrict__ fshift,
                                  float* __restrict__ pooled)
{
    const int t = threadIdx.x, lane = t & 31, wave = t >> 5;
    const int q = blockIdx.x * 8 + wave;
    const int b = q / Mm;

    const float qx = new_xyz[(size_t)q * 3 + 0];
    const float qy = new_xyz[(size_t)q * 3 + 1];
    const float qz = new_xyz[(size_t)q * 3 + 2];

    const int c0 = lane * 2, c1 = c0 + 1;
    const float w00 = Wx[c0],        w01 = Wx[c1];
    const float w10 = Wx[C0 + c0],   w11 = Wx[C0 + c1];
    const float w20 = Wx[2*C0 + c0], w21 = Wx[2*C0 + c1];
    const float xs0 = xscale[c0], xh0 = xshift[c0];
    const float xs1 = xscale[c1], xh1 = xshift[c1];
    const float fs0 = fscale[c0], fh0 = fshift[c0];
    const float fs1 = fscale[c1], fh1 = fshift[c1];

    const int myidx = idx[(size_t)q * Kk + lane];
    const float* __restrict__ xb = xyz + (size_t)b * Nn * 3;

    float mx0 = 0.f, mx1 = 0.f;   // ReLU outputs are >= 0
    for (int k = 0; k < Kk; ++k) {
        const int i = __shfl(myidx, k, 32);
        const float dx = xb[i * 3 + 0] - qx;
        const float dy = xb[i * 3 + 1] - qy;
        const float dz = xb[i * 3 + 2] - qz;
        const float2 f = *(const float2*)(Y + ((size_t)(b * Nn + i)) * C0 + c0);
        float v0 = (dx * w00 + dy * w10 + dz * w20) * xs0 + xh0 + (f.x * fs0 + fh0);
        float v1 = (dx * w01 + dy * w11 + dz * w21) * xs1 + xh1 + (f.y * fs1 + fh1);
        v0 = fmaxf(v0, 0.f);
        v1 = fmaxf(v1, 0.f);
        mx0 = fmaxf(mx0, v0);
        mx1 = fmaxf(mx1, v1);
    }
    float2 o; o.x = mx0; o.y = mx1;
    *(float2*)(pooled + (size_t)q * C0 + c0) = o;
}

// Final: out = relu(bn2(H2)) * (cnt>0)
__global__ void output_kernel(const float* __restrict__ H2,
                              const float* __restrict__ sc,
                              const float* __restrict__ sh,
                              const int* __restrict__ cnt,
                              float* __restrict__ out)
{
    const int t = blockIdx.x * 256 + threadIdx.x;    // < B*M*C2 exactly
    const int c = t & (C2 - 1);
    const int q = t >> 8;
    float v = H2[t] * sc[c] + sh[c];
    v = v > 0.f ? v : 0.f;
    out[t] = (cnt[q] > 0) ? v : 0.f;
}

// =====================================================================
extern "C" void kernel_launch(void* const* d_in, const int* in_sizes, int n_in,
                              void* d_out, int out_size, void* d_ws, size_t ws_size,
                              hipStream_t stream)
{
    const float* xyz     = (const float*)d_in[0];
    const float* new_xyz = (const float*)d_in[1];
    const float* feats   = (const float*)d_in[2];
    const float* W_feat  = (const float*)d_in[3];
    const float* g_feat  = (const float*)d_in[4];
    const float* b_feat  = (const float*)d_in[5];
    const float* W_xyz   = (const float*)d_in[6];
    const float* g_xyz   = (const float*)d_in[7];
    const float* b_xyz   = (const float*)d_in[8];
    const float* W1      = (const float*)d_in[9];
    const float* g1      = (const float*)d_in[10];
    const float* b1      = (const float*)d_in[11];
    const float* W2      = (const float*)d_in[12];
    const float* g2      = (const float*)d_in[13];
    const float* b2      = (const float*)d_in[14];
    float* out = (float*)d_out;

    float* ws   = (float*)d_ws;
    float* Y    = ws + WS_Y;
    float* H1   = ws + WS_H1;
    float* H2   = ws + WS_H2;
    float* POOL = ws + WS_POOL;
    int*   IDX  = (int*)(ws + WS_IDX);
    int*   CNT  = (int*)(ws + WS_CNT);
    float* ST   = ws + WS_STATS;

    // zero the accumulator region (sums + moments)
    (void)hipMemsetAsync(ST, 0, 928 * sizeof(float), stream);

    // 1) feat encoder GEMM (WMMA bf16) + BN stats
    gemm_bn_stats_kernel<false, CI, C0><<<(Bb * Nn / 16) * (C0 / 16) / 8, 256, 0, stream>>>(
        feats, W_feat, Y, ST + SF1, ST + SF2, nullptr, nullptr);
    bn_finalize_kernel<<<1, 64, 0, stream>>>(ST + SF1, ST + SF2, g_feat, b_feat,
                                             1.0f / (float)(Bb * Nn), ST + FSC, ST + FSH, C0);

    // 2) ordered first-K ball query (TDM double-buffered LDS staging)
    ball_query_kernel<<<Bb * Mm / 8, 256, 0, stream>>>(xyz, new_xyz, IDX, CNT);

    // 3) grouped-xyz BN stats via 9 moments
    xyz_moments_kernel<<<Bb * Mm * Kk / 256, 256, 0, stream>>>(xyz, new_xyz, IDX, ST + SMOM);
    xyz_bn_finalize_kernel<<<1, 64, 0, stream>>>(ST + SMOM, W_xyz, g_xyz, b_xyz,
                                                 ST + XSC, ST + XSH);

    // 4) fused gather + BN + add + ReLU + max-pool
    group_pool_kernel<<<Bb * Mm / 8, 256, 0, stream>>>(xyz, new_xyz, Y, IDX, W_xyz,
                                                       ST + XSC, ST + XSH,
                                                       ST + FSC, ST + FSH, POOL);

    // 5) MLP1 GEMM (WMMA) + BN1 stats
    gemm_bn_stats_kernel<false, C0, C1><<<(Bb * Mm / 16) * (C1 / 16) / 8, 256, 0, stream>>>(
        POOL, W1, H1, ST + S11, ST + S12, nullptr, nullptr);
    bn_finalize_kernel<<<1, 128, 0, stream>>>(ST + S11, ST + S12, g1, b1,
                                              1.0f / (float)(Bb * Mm), ST + SC1, ST + SH1, C1);

    // 6) MLP2 GEMM (WMMA, fused BN1+ReLU on A-load) + BN2 stats
    gemm_bn_stats_kernel<true, C1, C2><<<(Bb * Mm / 16) * (C2 / 16) / 8, 256, 0, stream>>>(
        H1, W2, H2, ST + S21, ST + S22, ST + SC1, ST + SH1);
    bn_finalize_kernel<<<1, 256, 0, stream>>>(ST + S21, ST + S22, g2, b2,
                                              1.0f / (float)(Bb * Mm), ST + SC2, ST + SH2, C2);

    // 7) final BN2 + ReLU + cnt mask
    output_kernel<<<Bb * Mm * C2 / 256, 256, 0, stream>>>(H2, ST + SC2, ST + SH2, CNT, out);
}
